// SpatialTemporalTensorAttention_39539468926993
// MI455X (gfx1250) — matrix operand order repaired and checked by reading, not compiled, not run on hardware
//
#include <hip/hip_runtime.h>

// ---------------------------------------------------------------------------
// MI455X (gfx1250, wave32) spatio-temporal attention.
// All matmuls via v_wmma_f32_16x16x32_bf16 (fp32 accum), softmax in fp32.
// GEMMs use ASYNCcnt-tracked global_load_async_to_lds_b128 double buffering.
// Shapes: T=64, S=128, B=8, E=512, H=8, Dh=64;  N = T*S*B = 65536 rows.
// ---------------------------------------------------------------------------

typedef __bf16 bf16_t;
typedef __attribute__((ext_vector_type(16))) __bf16    v16bf;
typedef __attribute__((ext_vector_type(8)))  float     v8f;
typedef __attribute__((ext_vector_type(4)))  unsigned  u32x4;

#define DEV __device__ __forceinline__

union FragU { u32x4 u[2]; v16bf v; };

// A fragment: 16x32 (MxK) bf16, row-major source with row stride ld (elems).
// ISA layout: lanes 0-15 = M rows, halves 0-7 -> K 0..7 (lane>=16: K 8..15),
// halves 8-15 -> K 16..23 (lane>=16: K 24..31).
DEV v16bf frag_a(const bf16_t* p0, size_t ld, int lane) {
  int m  = lane & 15;
  int hi = (lane >> 4) << 3;                 // 0 or 8
  const bf16_t* p = p0 + (size_t)m * ld + hi;
  FragU x;
  x.u[0] = *(const u32x4*)(p);
  x.u[1] = *(const u32x4*)(p + 16);
  return x.v;
}

// B fragment: 32x16 (KxN) bf16 where column n of B = row n of W (row-major,
// stride ld). Same K layout as A: halves 0-7 -> K 0..7 (lane>=16: K 8..15),
// halves 8-15 -> K 16..23 (lane>=16: K 24..31).
DEV v16bf frag_b(const bf16_t* p0, size_t ld, int lane) {
  int n  = lane & 15;
  int hi = (lane >> 4) << 3;                 // 0 or 8
  const bf16_t* p = p0 + (size_t)n * ld + hi;
  FragU x;
  x.u[0] = *(const u32x4*)(p);
  x.u[1] = *(const u32x4*)(p + 16);
  return x.v;
}

DEV v8f wmma_bf16(v16bf a, v16bf b, v8f c) {
  return __builtin_amdgcn_wmma_f32_16x16x32_bf16(
      false, a, false, b, (short)0, c, false, false);
}

// Async 16-byte global -> LDS copy, tracked by ASYNCcnt.
DEV void async_b128(const bf16_t* gsrc, bf16_t* ldst) {
  unsigned l = (unsigned)(unsigned long long)ldst;       // LDS byte offset
  asm volatile("global_load_async_to_lds_b128 %0, %1, off"
               :: "v"(l), "v"((unsigned long long)gsrc)
               : "memory");
}

DEV void wait_async0() {
  asm volatile("s_wait_asynccnt 0x0" ::: "memory");
}

// ---------------------------------------------------------------------------
// fp32 -> bf16 conversion (bandwidth trivial)
// ---------------------------------------------------------------------------
__global__ void __launch_bounds__(256)
cvt_f32_bf16(const float* __restrict__ in, bf16_t* __restrict__ out, long n) {
  long i = ((long)blockIdx.x * 256 + threadIdx.x) * 4;
  if (i + 3 < n) {
    out[i + 0] = (bf16_t)in[i + 0];
    out[i + 1] = (bf16_t)in[i + 1];
    out[i + 2] = (bf16_t)in[i + 2];
    out[i + 3] = (bf16_t)in[i + 3];
  }
}

// ---------------------------------------------------------------------------
// GEMM: out[N x F] = A[N x K](bf16) * W[F x K]^T(bf16) + bias(f32)
// Block: 128 rows x 64 cols, 256 threads (8 waves in 4x2), 2x2 tiles/wave.
// K-tiles of 32 staged into LDS via async DMA, double buffered; LDS rows
// padded to 40 elems (80 B) so b128 fragment reads spread across banks.
// OUTF32: write fp32 (final output); REMAP: (s,t,b) -> (t,s,b) row remap.
// ---------------------------------------------------------------------------
#define TILE_ELEMS ((128 + 64) * 40)

template <int OUTF32, int REMAP>
__global__ void __launch_bounds__(256)
gemm_bias_kernel(const bf16_t* __restrict__ A, const bf16_t* __restrict__ W,
                 const float* __restrict__ bias, bf16_t* __restrict__ outB,
                 float* __restrict__ outF, int K, int F) {
  __shared__ __align__(16) bf16_t lds[2][TILE_ELEMS];

  const int tid  = threadIdx.x;
  const int lane = tid & 31;
  const int wave = tid >> 5;
  const int wm = wave >> 1;                  // 0..3
  const int wn = wave & 1;                   // 0..1
  const long rowB = (long)blockIdx.y * 128;
  const long colB = (long)blockIdx.x * 64;

  // Stage one 128x32 A tile + 64x32 B tile (k-offset kk) into lds[buf].
  // Chunks of 8 bf16 (16 B): A has 512 chunks (2/thread), B has 256 (1/thread).
  auto stage = [&](int buf, int kk) {
    bf16_t* dA = &lds[buf][0];
    bf16_t* dB = &lds[buf][128 * 40];
#pragma unroll
    for (int t = 0; t < 2; ++t) {
      int c = tid + t * 256;
      int r = c >> 2, cc = (c & 3) * 8;
      async_b128(A + (size_t)(rowB + r) * K + kk + cc, dA + r * 40 + cc);
    }
    {
      int r = tid >> 2, cc = (tid & 3) * 8;
      async_b128(W + (size_t)(colB + r) * K + kk + cc, dB + r * 40 + cc);
    }
  };

  stage(0, 0);
  wait_async0();
  __syncthreads();

  v8f acc[2][2] = {};
  const int nsteps = K >> 5;
  for (int it = 0; it < nsteps; ++it) {
    const int cur = it & 1;
    if (it + 1 < nsteps) stage(cur ^ 1, (it + 1) << 5);   // DMA overlaps WMMA

    const bf16_t* tA = &lds[cur][0];
    const bf16_t* tB = &lds[cur][128 * 40];
    v16bf a0 = frag_a(tA + (wm * 32 + 0)  * 40, 40, lane);
    v16bf a1 = frag_a(tA + (wm * 32 + 16) * 40, 40, lane);
    v16bf b0 = frag_b(tB + (wn * 32 + 0)  * 40, 40, lane);
    v16bf b1 = frag_b(tB + (wn * 32 + 16) * 40, 40, lane);
    acc[0][0] = wmma_bf16(a0, b0, acc[0][0]);
    acc[0][1] = wmma_bf16(a0, b1, acc[0][1]);
    acc[1][0] = wmma_bf16(a1, b0, acc[1][0]);
    acc[1][1] = wmma_bf16(a1, b1, acc[1][1]);

    if (it + 1 < nsteps) {
      wait_async0();        // next tile landed (per-wave), readers done via barrier
      __syncthreads();
    }
  }

  const int nl = lane & 15;
  const int hi = (lane >> 4) * 8;            // C layout: M = r + hi, N = nl
  for (int i = 0; i < 2; ++i)
    for (int j = 0; j < 2; ++j) {
      long col = colB + wn * 32 + j * 16 + nl;
      float bv = bias[col];
      for (int r = 0; r < 8; ++r) {
        long g = rowB + wm * 32 + i * 16 + r + hi;
        float v = acc[i][j][r] + bv;
        if (OUTF32) {
          long o = g;
          if (REMAP) {                        // g=(s*64+t)*8+b -> (t*128+s)*8+b
            long s = g >> 9, rr = g & 511, t = rr >> 3, b = rr & 7;
            o = (t << 10) + (s << 3) + b;
          }
          outF[(size_t)o * F + col] = v;
        } else {
          outB[(size_t)g * F + col] = (bf16_t)v;
        }
      }
    }
}

// ---------------------------------------------------------------------------
// Temporal attention: one block per (h, n), n in [0,1024) = s*8+b.
// q/k/v from qkv[N x 1536]: row t*1024+n, cols [h*64), [512+h*64), [1024+h*64).
// Output bf16 directly in xs layout: row s*512 + t*8 + b.
// ---------------------------------------------------------------------------
__global__ void __launch_bounds__(128)
temporal_attn_kernel(const bf16_t* __restrict__ qkv, bf16_t* __restrict__ xsb,
                     float scale) {
  const int h = blockIdx.x, n = blockIdx.y;
  const int tid = threadIdx.x, lane = tid & 31, wave = tid >> 5;  // 4 waves

  extern __shared__ char smem[];
  float*  sc = (float*)smem;                       // 64 x 65 (f32)
  bf16_t* pm = (bf16_t*)(smem + 64 * 65 * 4);      // 64 x 72 (bf16 P)
  bf16_t* vt = pm + 64 * 72;                       // 64 x 72 (bf16 V^T: [d][u])

  const size_t ld = (size_t)1024 * 1536;           // stride between t rows
  const bf16_t* qb = qkv + (size_t)n * 1536 + h * 64;
  const bf16_t* kb = qb + 512;
  const bf16_t* vb = qb + 1024;

  // scores: wave handles query rows [wave*16, +16), all 64 keys (4 tiles)
  v8f acc[4] = {};
  for (int kk = 0; kk < 64; kk += 32) {
    v16bf a = frag_a(qb + (size_t)(wave * 16) * ld + kk, ld, lane);
    for (int j = 0; j < 4; ++j) {
      v16bf b = frag_b(kb + (size_t)(j * 16) * ld + kk, ld, lane);
      acc[j] = wmma_bf16(a, b, acc[j]);
    }
  }
  const int nl = lane & 15, hi = (lane >> 4) * 8;
  for (int j = 0; j < 4; ++j)
    for (int r = 0; r < 8; ++r)
      sc[(wave * 16 + r + hi) * 65 + j * 16 + nl] = acc[j][r] * scale;

  // stage V transposed into LDS: vt[d][u]
  {
    int u = tid >> 1, d0 = (tid & 1) * 32;
    const bf16_t* src = vb + (size_t)u * ld + d0;
    for (int i = 0; i < 32; ++i) vt[(d0 + i) * 72 + u] = src[i];
  }
  __syncthreads();

  if (tid < 64) {                                  // row-wise softmax
    int t = tid;
    float m = -1e30f;
    for (int u = 0; u < 64; ++u) m = fmaxf(m, sc[t * 65 + u]);
    float s = 0.f;
    for (int u = 0; u < 64; ++u) s += __expf(sc[t * 65 + u] - m);
    float inv = 1.0f / s;
    for (int u = 0; u < 64; ++u)
      pm[t * 72 + u] = (bf16_t)(__expf(sc[t * 65 + u] - m) * inv);
  }
  __syncthreads();

  // O = P @ V : 16 x 64 per wave, K = 64 keys
  v8f o[4] = {};
  for (int kk = 0; kk < 64; kk += 32) {
    v16bf a = frag_a(pm + (size_t)(wave * 16) * 72 + kk, 72, lane);
    for (int j = 0; j < 4; ++j) {
      v16bf b = frag_b(vt + (size_t)(j * 16) * 72 + kk, 72, lane);
      o[j] = wmma_bf16(a, b, o[j]);
    }
  }
  const int s_ = n >> 3, b_ = n & 7;
  for (int j = 0; j < 4; ++j)
    for (int r = 0; r < 8; ++r) {
      int t = wave * 16 + r + hi;
      xsb[((size_t)s_ * 512 + t * 8 + b_) * 512 + h * 64 + j * 16 + nl] =
          (bf16_t)o[j][r];
    }
}

// ---------------------------------------------------------------------------
// Spatial attention: one block per (h, n), n in [0,512) = t*8+b; S=128.
// Inter q/k in qki[N x 1024] (row s*512+n), self q/k in qks (same layout).
// Values = xsb rows. Self branch contributes only the diagonal.
// ---------------------------------------------------------------------------
__global__ void __launch_bounds__(256)
spatial_attn_kernel(const bf16_t* __restrict__ qki,
                    const bf16_t* __restrict__ qks,
                    const bf16_t* __restrict__ xsb,
                    bf16_t* __restrict__ osb, float scale) {
  const int h = blockIdx.x, n = blockIdx.y;
  const int tid = threadIdx.x, lane = tid & 31, wave = tid >> 5;  // 8 waves

  extern __shared__ char smem[];
  float*  sc = (float*)smem;                       // 128 x 129 (f32)
  bf16_t* pm = (bf16_t*)(smem + 128 * 129 * 4);    // 128 x 136 (bf16 P)
  bf16_t* vt = pm + 128 * 136;                     // 64 x 136 (bf16 V^T)

  const size_t ld = (size_t)512 * 1024;            // stride between s rows
  const bf16_t* qb = qki + (size_t)n * 1024 + h * 64;
  const bf16_t* kb = qb + 512;

  // inter scores: wave handles rows [wave*16, +16), 128 keys (8 tiles)
  v8f acc[8] = {};
  for (int kk = 0; kk < 64; kk += 32) {
    v16bf a = frag_a(qb + (size_t)(wave * 16) * ld + kk, ld, lane);
    for (int j = 0; j < 8; ++j) {
      v16bf b = frag_b(kb + (size_t)(j * 16) * ld + kk, ld, lane);
      acc[j] = wmma_bf16(a, b, acc[j]);
    }
  }
  const int nl = lane & 15, hi = (lane >> 4) * 8;
  for (int j = 0; j < 8; ++j)
    for (int r = 0; r < 8; ++r)
      sc[(wave * 16 + r + hi) * 129 + j * 16 + nl] = acc[j][r] * scale;

  // stage V transposed: keys r in [0,128), vt[d][r]
  {
    int r = tid >> 1, d0 = (tid & 1) * 32;
    const bf16_t* src = xsb + ((size_t)r * 512 + n) * 512 + h * 64 + d0;
    for (int i = 0; i < 32; ++i) vt[(d0 + i) * 136 + r] = src[i];
  }
  __syncthreads();

  if (tid < 128) {
    int s = tid;
    // self-branch: only the diagonal survives the (1-I)/I blend
    const bf16_t* qs = qks + ((size_t)s * 512 + n) * 1024 + h * 64;
    const bf16_t* ks = qs + 512;
    float dot = 0.f;
    for (int d = 0; d < 64; ++d) dot += (float)qs[d] * (float)ks[d];
    sc[s * 129 + s] = dot * scale;

    float m = -1e30f;
    for (int r = 0; r < 128; ++r) m = fmaxf(m, sc[s * 129 + r]);
    float ssum = 0.f;
    for (int r = 0; r < 128; ++r) ssum += __expf(sc[s * 129 + r] - m);
    float inv = 1.0f / ssum;
    for (int r = 0; r < 128; ++r)
      pm[s * 136 + r] = (bf16_t)(__expf(sc[s * 129 + r] - m) * inv);
  }
  __syncthreads();

  // O = P @ V : 16 x 64 per wave, K = 128 keys
  v8f o[4] = {};
  for (int kk = 0; kk < 128; kk += 32) {
    v16bf a = frag_a(pm + (size_t)(wave * 16) * 136 + kk, 136, lane);
    for (int j = 0; j < 4; ++j) {
      v16bf b = frag_b(vt + (size_t)(j * 16) * 136 + kk, 136, lane);
      o[j] = wmma_bf16(a, b, o[j]);
    }
  }
  for (int j = 0; j < 4; ++j)
    for (int r = 0; r < 8; ++r) {
      int s = wave * 16 + r + hi;
      osb[((size_t)s * 512 + n) * 512 + h * 64 + j * 16 + nl] =
          (bf16_t)o[j][r];
    }
}

// ---------------------------------------------------------------------------
// Host launcher
// ---------------------------------------------------------------------------
extern "C" void kernel_launch(void* const* d_in, const int* in_sizes, int n_in,
                              void* d_out, int out_size, void* d_ws,
                              size_t ws_size, hipStream_t stream) {
  (void)in_sizes; (void)n_in; (void)out_size; (void)ws_size;
  const float* x   = (const float*)d_in[0];
  const float* Wt  = (const float*)d_in[1];
  const float* bt  = (const float*)d_in[2];
  const float* Ws  = (const float*)d_in[3];
  const float* bs  = (const float*)d_in[4];
  const float* Wss = (const float*)d_in[5];
  const float* bss = (const float*)d_in[6];
  const float* Wo  = (const float*)d_in[7];
  const float* bo  = (const float*)d_in[8];
  float* y = (float*)d_out;

  const long N = 65536;                            // T*S*B

  char* w = (char*)d_ws;
  auto carve = [&](size_t bytes) {
    char* p = w;
    w += (bytes + 255) & ~(size_t)255;
    return p;
  };
  bf16_t* xb   = (bf16_t*)carve((size_t)N * 512 * 2);
  bf16_t* Wtb  = (bf16_t*)carve((size_t)1536 * 512 * 2);
  bf16_t* Wsb  = (bf16_t*)carve((size_t)1024 * 512 * 2);
  bf16_t* Wssb = (bf16_t*)carve((size_t)1024 * 512 * 2);
  bf16_t* Wob  = (bf16_t*)carve((size_t)512 * 512 * 2);
  bf16_t* qkvt = (bf16_t*)carve((size_t)N * 1536 * 2);
  bf16_t* xsb  = (bf16_t*)carve((size_t)N * 512 * 2);
  bf16_t* qki  = (bf16_t*)carve((size_t)N * 1024 * 2);
  bf16_t* qks  = (bf16_t*)carve((size_t)N * 1024 * 2);
  bf16_t* osb  = (bf16_t*)carve((size_t)N * 512 * 2);

  // fp32 -> bf16
  cvt_f32_bf16<<<dim3((unsigned)(N * 512 / 1024)), dim3(256), 0, stream>>>(x, xb, N * 512);
  cvt_f32_bf16<<<dim3(1536 * 512 / 1024), dim3(256), 0, stream>>>(Wt, Wtb, 1536l * 512);
  cvt_f32_bf16<<<dim3(1024 * 512 / 1024), dim3(256), 0, stream>>>(Ws, Wsb, 1024l * 512);
  cvt_f32_bf16<<<dim3(1024 * 512 / 1024), dim3(256), 0, stream>>>(Wss, Wssb, 1024l * 512);
  cvt_f32_bf16<<<dim3(512 * 512 / 1024), dim3(256), 0, stream>>>(Wo, Wob, 512l * 512);

  // temporal qkv projection: (65536 x 512) x (1536 x 512)^T
  gemm_bias_kernel<0, 0><<<dim3(1536 / 64, N / 128), dim3(256), 0, stream>>>(
      xb, Wtb, bt, qkvt, nullptr, 512, 1536);

  // temporal attention (writes xs layout)
  temporal_attn_kernel<<<dim3(8, 1024), dim3(128),
                         64 * 65 * 4 + 2 * 64 * 72 * 2, stream>>>(
      qkvt, xsb, 0.125f);

  // spatial inter + self projections
  gemm_bias_kernel<0, 0><<<dim3(1024 / 64, N / 128), dim3(256), 0, stream>>>(
      xsb, Wsb, bs, qki, nullptr, 512, 1024);
  gemm_bias_kernel<0, 0><<<dim3(1024 / 64, N / 128), dim3(256), 0, stream>>>(
      xsb, Wssb, bss, qks, nullptr, 512, 1024);

  // spatial attention (writes (S,T,B,E) layout)
  spatial_attn_kernel<<<dim3(8, 512), dim3(256),
                        128 * 129 * 4 + 128 * 136 * 2 + 64 * 136 * 2, stream>>>(
      qki, qks, xsb, osb, 0.125f);

  // output projection with (s,t,b) -> (t,s,b) remap, fp32 out
  gemm_bias_kernel<1, 1><<<dim3(512 / 64, N / 128), dim3(256), 0, stream>>>(
      osb, Wob, bo, nullptr, y, 512, 512);
}
